// branch_layer_72988674228877
// MI455X (gfx1250) — matrix operand order repaired and checked by reading, not compiled, not run on hardware
//
#include <hip/hip_runtime.h>

// ---------------------------------------------------------------------------
// 8-branch fused MLP mixture for MI455X (gfx1250, wave32, WMMA bf16).
//   out[b,o] = sum_n prob[b,n] * relu( relu(x[b,:] @ W1[n] + b1[n]) @ W2[n] + b2[n] )
// Memory-bound (0.55 TFLOP vs ~200 MB min HBM traffic): fuse both GEMMs
// through LDS, convert f32->bf16 in-register, keep the probability-weighted
// output in VGPRs, write output exactly once. Each wave computes BOTH
// 16-row m-subtiles so every global weight (B) fragment feeds 2 WMMAs.
// ---------------------------------------------------------------------------

#define NBR 8
#define INS 1024
#define HID 2048
#define OUTS 1024
#define BATCH 8192
#define MT 32            // batch rows per workgroup (2 m-subtiles of 16)
#define NTHREADS 512     // 16 waves = 16 n-groups

typedef __attribute__((ext_vector_type(16))) __bf16 v16bf;
typedef __attribute__((ext_vector_type(8)))  float  v8f;

// A-fragment (16x32 bf16, ISA 7.12.2 "16-bit A-Matrix 16x32"):
// lane<16: row=lane, k = 0..15 (contiguous, matches B fragment)
// lane>=16: row=lane-16, k = 16..31
__device__ __forceinline__ v16bf load_frag_a(const __bf16* p, int hi) {
  const __bf16* q = p + hi * 16;
  union { uint4 u[2]; v16bf v; } r;
  r.u[0] = *(const uint4*)(q);        // k .. k+7      (ds_load_b128)
  r.u[1] = *(const uint4*)(q + 8);    // k+8 .. k+15   (ds_load_b128)
  return r.v;
}

// B-fragment (32x16 bf16): lane<16 holds K=0..15 (2/VGPR), lane>=16 K=16..31,
// column n = lane&15. Loaded from row-major f32 weights [K][ldn]; pairs of
// converts fuse into v_cvt_pk_bf16_f32.
__device__ __forceinline__ v16bf load_frag_b(const float* W, int ldn,
                                             int kbase, int col, int hi) {
  const float* p = W + (size_t)(kbase + hi * 16) * ldn + col;
  v16bf f;
#pragma unroll
  for (int i = 0; i < 16; ++i) f[i] = (__bf16)p[(size_t)i * ldn];
  return f;
}

__device__ __forceinline__ v8f splat8(float s) {
  v8f r;
#pragma unroll
  for (int i = 0; i < 8; ++i) r[i] = s;
  return r;
}

__global__ void __launch_bounds__(NTHREADS)
branch_mlp_wmma_kernel(const float* __restrict__ x,
                       const float* __restrict__ prob,
                       const float* __restrict__ W1,
                       const float* __restrict__ b1,
                       const float* __restrict__ W2,
                       const float* __restrict__ b2,
                       float* __restrict__ out) {
  extern __shared__ char smem[];
  __bf16* xs = (__bf16*)smem;            // [MT][INS]  x tile (bf16)   64 KB
  __bf16* hs = xs + MT * INS;            // [MT][1024] hidden chunk    64 KB
  float*  ps = (float*)(hs + MT * 1024); // [MT] per-row probabilities

  const int tid  = threadIdx.x;
  const int wave = tid >> 5;             // 0..15 : n-group (64 cols each)
  const int lane = tid & 31;
  const int l16  = lane & 15;
  const int hi   = lane >> 4;
  const int rowBase = blockIdx.x * MT;

  // Stage + convert the x tile once (float4 loads, packed bf16 stores).
  {
    const float* xbase = x + (size_t)rowBase * INS;  // 128 KB contiguous block
    for (int i = tid * 4; i < MT * INS; i += NTHREADS * 4) {
      float4 f = *(const float4*)(xbase + i);
      unsigned lo = __builtin_bit_cast(unsigned,
          (float)0);  // placeholder to keep types simple
      (void)lo;
      __bf16 b0 = (__bf16)f.x, b1v = (__bf16)f.y,
             b2v = (__bf16)f.z, b3 = (__bf16)f.w;
      union { __bf16 h[4]; unsigned u[2]; } pk;
      pk.h[0] = b0; pk.h[1] = b1v; pk.h[2] = b2v; pk.h[3] = b3;
      *(unsigned*)(xs + i)     = pk.u[0];            // ds_store_b64-able
      *(unsigned*)(xs + i + 2) = pk.u[1];
    }
  }

  const v8f vz = {};
  v8f acc[2][4];                         // weighted output: 2 m-sub x 4 n-tiles
#pragma unroll
  for (int s = 0; s < 2; ++s)
#pragma unroll
    for (int t = 0; t < 4; ++t) acc[s][t] = vz;

  const __bf16* xrow0 = xs + l16 * INS;          // m-subtile 0 (rows 0..15)
  const __bf16* xrow1 = xs + (16 + l16) * INS;   // m-subtile 1 (rows 16..31)
  const __bf16* hrow0 = hs + l16 * 1024;
  const __bf16* hrow1 = hs + (16 + l16) * 1024;

  for (int br = 0; br < NBR; ++br) {
    __syncthreads();                     // prev branch done with ps/hs
    if (tid < MT) ps[tid] = prob[(size_t)(rowBase + tid) * NBR + br];
    const float* W1b = W1 + (size_t)br * INS * HID;
    const float* W2b = W2 + (size_t)br * HID * OUTS;
    const float* b1b = b1 + br * HID;
    const float* b2b = b2 + br * OUTS;

    // o-accumulators initialized with b2 bias (n-only => same in all 8 VGPRs)
    v8f oacc[2][4];
#pragma unroll
    for (int t = 0; t < 4; ++t) {
      v8f bi = splat8(b2b[wave * 64 + t * 16 + l16]);
      oacc[0][t] = bi;
      oacc[1][t] = bi;
    }

    // hidden dimension processed in 2 chunks of 1024 through the hs buffer
    for (int ch = 0; ch < 2; ++ch) {
      __syncthreads();                   // hs free for writing
      // ---- GEMM1: h[32, chunk] = relu(x @ W1 + b1) -> hs (bf16) ----
      for (int t = 0; t < 4; ++t) {
        const int ncl = wave * 64 + t * 16 + l16;  // col inside chunk
        const int ncg = ch * 1024 + ncl;           // col inside HID
        v8f d0 = splat8(b1b[ncg]);                 // bias folded into C
        v8f d1 = d0;
        for (int ks = 0; ks < INS / 32; ++ks) {
          v16bf bf = load_frag_b(W1b, HID, ks * 32, ncg, hi);   // shared
          v16bf a0 = load_frag_a(xrow0 + ks * 32, hi);
          v16bf a1 = load_frag_a(xrow1 + ks * 32, hi);
          d0 = __builtin_amdgcn_wmma_f32_16x16x32_bf16(
                 false, a0, false, bf, (short)0, d0, false, false);
          d1 = __builtin_amdgcn_wmma_f32_16x16x32_bf16(
                 false, a1, false, bf, (short)0, d1, false, false);
        }
#pragma unroll
        for (int v = 0; v < 8; ++v) {
          float h0 = __builtin_fmaxf(d0[v], 0.f);
          float h1 = __builtin_fmaxf(d1[v], 0.f);
          hs[(v + 8 * hi) * 1024 + ncl]      = (__bf16)h0;
          hs[(16 + v + 8 * hi) * 1024 + ncl] = (__bf16)h1;
        }
      }
      __syncthreads();                   // hs ready for reading
      // ---- GEMM2 partial: oacc += h_chunk @ W2[chunk rows] ----
#pragma unroll
      for (int t = 0; t < 4; ++t) {
        const int ncol = wave * 64 + t * 16 + l16; // output col
        v8f d0 = oacc[0][t];
        v8f d1 = oacc[1][t];
        for (int ks = 0; ks < 1024 / 32; ++ks) {
          v16bf bf = load_frag_b(W2b, OUTS, ch * 1024 + ks * 32, ncol, hi);
          v16bf a0 = load_frag_a(hrow0 + ks * 32, hi);
          v16bf a1 = load_frag_a(hrow1 + ks * 32, hi);
          d0 = __builtin_amdgcn_wmma_f32_16x16x32_bf16(
                 false, a0, false, bf, (short)0, d0, false, false);
          d1 = __builtin_amdgcn_wmma_f32_16x16x32_bf16(
                 false, a1, false, bf, (short)0, d1, false, false);
        }
        oacc[0][t] = d0;
        oacc[1][t] = d1;
      }
    }

    // epilogue: relu(o) (bias already in C) weighted by per-row probability
#pragma unroll
    for (int s = 0; s < 2; ++s)
#pragma unroll
      for (int t = 0; t < 4; ++t)
#pragma unroll
        for (int v = 0; v < 8; ++v) {
          float ov = __builtin_fmaxf(oacc[s][t][v], 0.f);
          acc[s][t][v] += ov * ps[s * 16 + v + 8 * hi];
        }
  }

  // single output write (64 B contiguous per 16-lane half)
#pragma unroll
  for (int s = 0; s < 2; ++s)
#pragma unroll
    for (int t = 0; t < 4; ++t) {
      const int ncol = wave * 64 + t * 16 + l16;
#pragma unroll
      for (int v = 0; v < 8; ++v) {
        const int m = rowBase + s * 16 + v + 8 * hi;
        out[(size_t)m * OUTS + ncol] = acc[s][t][v];
      }
    }
}

extern "C" void kernel_launch(void* const* d_in, const int* in_sizes, int n_in,
                              void* d_out, int out_size, void* d_ws, size_t ws_size,
                              hipStream_t stream) {
  const float* x    = (const float*)d_in[0];
  const float* prob = (const float*)d_in[1];
  const float* W1   = (const float*)d_in[2];
  const float* b1   = (const float*)d_in[3];
  const float* W2   = (const float*)d_in[4];
  const float* b2   = (const float*)d_in[5];
  float* out = (float*)d_out;

  const size_t smem_bytes = (size_t)MT * INS * 2 + (size_t)MT * 1024 * 2 + MT * 4;
  hipFuncSetAttribute((const void*)branch_mlp_wmma_kernel,
                      hipFuncAttributeMaxDynamicSharedMemorySize,
                      (int)smem_bytes);

  dim3 grid(BATCH / MT);   // 256 workgroups
  dim3 block(NTHREADS);    // 16 waves (wave32)
  branch_mlp_wmma_kernel<<<grid, block, smem_bytes, stream>>>(
      x, prob, W1, b1, W2, b2, out);
}
